// ProblematicAttention_77756087927150
// MI455X (gfx1250) — matrix operand order repaired and checked by reading, not compiled, not run on hardware
//
#include <hip/hip_runtime.h>
#include <hip/hip_bf16.h>
#include <stdint.h>

// Problem constants (from the reference)
#define BB 2
#define SS 2048
#define DD 512
#define HH 8
#define DH 64

typedef __attribute__((ext_vector_type(16))) __bf16 v16bf;
typedef __attribute__((ext_vector_type(8)))  float  v8f;
typedef __attribute__((ext_vector_type(4)))  int    v4i;

union Frag16 { uint4 q[2]; v16bf v; };

// gfx1250 async global->LDS copy path (ASYNCcnt), guarded for toolchain support
#if defined(__has_builtin)
#if __has_builtin(__builtin_amdgcn_global_load_async_to_lds_b128)
#define HAVE_ASYNC_LDS 1
#endif
#endif
#ifndef HAVE_ASYNC_LDS
#define HAVE_ASYNC_LDS 0
#endif

static __device__ inline __bf16 f2bf(float f) {
    union { float f; unsigned u; } in; in.f = f;
    unsigned u = in.u;
    u += 0x7FFFu + ((u >> 16) & 1u);          // round-to-nearest-even
    union { unsigned short s; __bf16 b; } out;
    out.s = (unsigned short)(u >> 16);
    return out.b;
}

// ---------------------------------------------------------------------------
// fp32 -> bf16 elementwise convert
// ---------------------------------------------------------------------------
__global__ void __launch_bounds__(256)
cvt_f32_bf16(const float* __restrict__ in, __bf16* __restrict__ out, int n) {
    int i = blockIdx.x * 256 + threadIdx.x;
    if (i < n) out[i] = f2bf(in[i]);
}

// ---------------------------------------------------------------------------
// Generic batched bf16 WMMA GEMM:  C[m][n] = scale * sum_k A[m][k]*B[n][k] + bias[n]
// One wave computes a 16x64 tile: 4 independent f32 accumulators so the four
// v_wmma_f32_16x16x32_bf16 per K-step have no RAW chain (fills the XDL pipe),
// and the A fragment is loaded once per K-step (4x reuse).
// A: [M,K] row-major bf16 (per batch z: +z*aBatch), lda elements
// B: [N,K] row-major bf16 (weights layout; per batch z: +z*bBatch), ldb elements
// Output index = (z/zH)*zs1 + (z%zH)*zs2 + (m/MD)*oms1 + (m%MD)*oms2
//              + (n/ND)*ons1 + (n%ND)*ons2        (f32 or bf16 elements)
// ---------------------------------------------------------------------------
__global__ void __launch_bounds__(128)
gemm_bf16_wmma(const __bf16* __restrict__ A, size_t aBatch, int lda,
               const __bf16* __restrict__ B, size_t bBatch, int ldb,
               const float*  __restrict__ bias, float scale,
               void* __restrict__ outPtr, int outBF16,
               int zH, size_t zs1, size_t zs2,
               int MD, size_t oms1, size_t oms2,
               int ND, size_t ons1, size_t ons2,
               int Msz, int Nsz, int Ksz)
{
    const int lane = threadIdx.x & 31;        // wave32
    const int wave = threadIdx.x >> 5;
    const int mTiles  = Msz >> 4;
    const int nTiles4 = Nsz >> 6;             // 64-wide wave tiles
    const int tile = blockIdx.x * 4 + wave;
    if (tile >= mTiles * nTiles4) return;
    const int mT = tile / nTiles4;
    const int nT = tile - mT * nTiles4;
    const int z  = blockIdx.y;
    const int r16 = lane & 15;
    const int hf  = lane >> 4;

    // A fragment (16x32 bf16): lane half 'hf' holds 16 contiguous K starting at hf*16
    const __bf16* aRow = A + (size_t)z * aBatch + (size_t)(mT * 16 + r16) * lda;
    // B fragments (32x16 bf16): lane half 'hf' holds 16 contiguous K starting at hf*16
    const __bf16* bBase = B + (size_t)z * bBatch + (size_t)(nT * 64 + r16) * ldb;
    const __bf16* bRow0 = bBase;
    const __bf16* bRow1 = bBase + (size_t)16 * ldb;
    const __bf16* bRow2 = bBase + (size_t)32 * ldb;
    const __bf16* bRow3 = bBase + (size_t)48 * ldb;

    v8f acc0 = {}, acc1 = {}, acc2 = {}, acc3 = {};
    for (int k = 0; k < Ksz; k += 32) {
        Frag16 fa, fb0, fb1, fb2, fb3;
        fa.q[0]  = *(const uint4*)(aRow  + k + hf * 16);
        fa.q[1]  = *(const uint4*)(aRow  + k + hf * 16 + 8);
        fb0.q[0] = *(const uint4*)(bRow0 + k + hf * 16);
        fb0.q[1] = *(const uint4*)(bRow0 + k + hf * 16 + 8);
        fb1.q[0] = *(const uint4*)(bRow1 + k + hf * 16);
        fb1.q[1] = *(const uint4*)(bRow1 + k + hf * 16 + 8);
        fb2.q[0] = *(const uint4*)(bRow2 + k + hf * 16);
        fb2.q[1] = *(const uint4*)(bRow2 + k + hf * 16 + 8);
        fb3.q[0] = *(const uint4*)(bRow3 + k + hf * 16);
        fb3.q[1] = *(const uint4*)(bRow3 + k + hf * 16 + 8);
        acc0 = __builtin_amdgcn_wmma_f32_16x16x32_bf16(false, fa.v, false, fb0.v, (short)0, acc0, false, false);
        acc1 = __builtin_amdgcn_wmma_f32_16x16x32_bf16(false, fa.v, false, fb1.v, (short)0, acc1, false, false);
        acc2 = __builtin_amdgcn_wmma_f32_16x16x32_bf16(false, fa.v, false, fb2.v, (short)0, acc2, false, false);
        acc3 = __builtin_amdgcn_wmma_f32_16x16x32_bf16(false, fa.v, false, fb3.v, (short)0, acc3, false, false);
    }

    // Epilogue: lane holds D rows (hf*8 + r), columns (nT*64 + j*16 + r16)
    const size_t zbase = (size_t)(z / zH) * zs1 + (size_t)(z % zH) * zs2;
    v8f accs[4] = { acc0, acc1, acc2, acc3 };
#pragma unroll
    for (int j = 0; j < 4; ++j) {
        const int nG = nT * 64 + j * 16 + r16;
        const float bv = bias ? bias[nG] : 0.0f;
        const size_t nOff = (size_t)(nG / ND) * ons1 + (size_t)(nG % ND) * ons2;
#pragma unroll
        for (int r = 0; r < 8; ++r) {
            const int mG = mT * 16 + hf * 8 + r;
            const size_t mOff = (size_t)(mG / MD) * oms1 + (size_t)(mG % MD) * oms2;
            const float v = accs[j][r] * scale + bv;
            const size_t off = zbase + mOff + nOff;
            if (outBF16) ((__bf16*)outPtr)[off] = f2bf(v);
            else         ((float*)outPtr)[off]  = v;
        }
    }
}

// ---------------------------------------------------------------------------
// Fused: A = softmax(scores); t = A*M; P = softmax(t)  -- one block per row.
// The M row is streamed into LDS with the gfx1250 async global->LDS engine
// (ASYNCcnt) while softmax #1 runs on the score row, overlapping the second
// HBM stream with the v_exp VALU work. P (bf16) is written IN PLACE over the
// first half of the f32 score row, so stage 4 reads it with lda = 2*S.
// ---------------------------------------------------------------------------
__device__ inline float blockMax(float v, float* sm) {
#pragma unroll
    for (int o = 16; o; o >>= 1) v = fmaxf(v, __shfl_xor(v, o, 32));
    __syncthreads();
    if ((threadIdx.x & 31) == 0) sm[threadIdx.x >> 5] = v;
    __syncthreads();
    float r = sm[0];
#pragma unroll
    for (int i = 1; i < 8; ++i) r = fmaxf(r, sm[i]);
    return r;
}

__device__ inline float blockSum(float v, float* sm) {
#pragma unroll
    for (int o = 16; o; o >>= 1) v += __shfl_xor(v, o, 32);
    __syncthreads();
    if ((threadIdx.x & 31) == 0) sm[threadIdx.x >> 5] = v;
    __syncthreads();
    float r = sm[0];
#pragma unroll
    for (int i = 1; i < 8; ++i) r += sm[i];
    return r;
}

__global__ void __launch_bounds__(256)
softmax_mask_softmax(float* __restrict__ scores, const float* __restrict__ Mmask)
{
    __shared__ float sm[8];
    __shared__ float mstage[SS];              // 8 KB of the 320 KB/WGP LDS
    const int row = blockIdx.x;               // 0..S-1
    const int z   = blockIdx.y;               // 0..B*H-1
    const int b   = z / HH;
    const int t   = threadIdx.x;

    float* srow = scores + ((size_t)z * SS + row) * SS;
    const float* mrow = Mmask + ((size_t)b * SS + row) * SS;
    __bf16* prow = (__bf16*)srow;             // in-place bf16 P row

#if HAVE_ASYNC_LDS
    // Kick off the async copy of the whole M row (2 x b128 per thread) first;
    // it proceeds in the background during softmax #1.
    __builtin_amdgcn_global_load_async_to_lds_b128(
        (v4i*)(mrow + t * 4),        (v4i*)(mstage + t * 4),        0, 0);
    __builtin_amdgcn_global_load_async_to_lds_b128(
        (v4i*)(mrow + 1024 + t * 4), (v4i*)(mstage + 1024 + t * 4), 0, 0);
#endif

    float sv[SS / 256];
#pragma unroll
    for (int i = 0; i < SS / 256; ++i) sv[i] = srow[i * 256 + t];

#if !HAVE_ASYNC_LDS
    // Fallback: synchronous staging of the M row into LDS.
#pragma unroll
    for (int i = 0; i < SS / 256; ++i) mstage[i * 256 + t] = mrow[i * 256 + t];
#endif

    // softmax #1 (overlaps with the in-flight async M copy)
    float mx = -3.402823466e38f;
#pragma unroll
    for (int i = 0; i < SS / 256; ++i) mx = fmaxf(mx, sv[i]);
    mx = blockMax(mx, sm);
    float sum = 0.f;
#pragma unroll
    for (int i = 0; i < SS / 256; ++i) sum += __expf(sv[i] - mx);
    sum = blockSum(sum, sm);
    const float inv = 1.0f / sum;

#if HAVE_ASYNC_LDS
    asm volatile("s_wait_asynccnt 0x0" ::: "memory");
#endif
    __syncthreads();                          // M row fully resident in LDS

#pragma unroll
    for (int i = 0; i < SS / 256; ++i)
        sv[i] = (__expf(sv[i] - mx) * inv) * mstage[i * 256 + t];   // A * M

    // softmax #2
    float mx2 = -3.402823466e38f;
#pragma unroll
    for (int i = 0; i < SS / 256; ++i) mx2 = fmaxf(mx2, sv[i]);
    mx2 = blockMax(mx2, sm);
    float sum2 = 0.f;
#pragma unroll
    for (int i = 0; i < SS / 256; ++i) sum2 += __expf(sv[i] - mx2);
    sum2 = blockSum(sum2, sm);
    const float inv2 = 1.0f / sum2;
#pragma unroll
    for (int i = 0; i < SS / 256; ++i)
        prow[i * 256 + t] = f2bf(__expf(sv[i] - mx2) * inv2);
}

// ---------------------------------------------------------------------------
// Host-side orchestration
// ---------------------------------------------------------------------------
static inline size_t alignUp(size_t v, size_t a) { return (v + a - 1) & ~(a - 1); }

extern "C" void kernel_launch(void* const* d_in, const int* in_sizes, int n_in,
                              void* d_out, int out_size, void* d_ws, size_t ws_size,
                              hipStream_t stream) {
    (void)in_sizes; (void)n_in; (void)out_size; (void)ws_size;

    const float* x        = (const float*)d_in[0];   // [B,S,D]
    const float* Mmask    = (const float*)d_in[1];   // [B,S,S]
    const float* in_w     = (const float*)d_in[2];   // [3D,D]
    const float* in_b     = (const float*)d_in[3];   // [3D]
    const float* v_w      = (const float*)d_in[4];   // [D,D]
    const float* v_b      = (const float*)d_in[5];   // [D]
    const float* out_w    = (const float*)d_in[6];   // [D,D]
    const float* out_b    = (const float*)d_in[7];   // [D]
    float* out            = (float*)d_out;           // [B,S,D]

    // Workspace layout (bytes)
    char* ws = (char*)d_ws;
    size_t off = 0;
    __bf16* xb  = (__bf16*)(ws + off); off = alignUp(off + (size_t)BB*SS*DD*2, 256);
    __bf16* wqk = (__bf16*)(ws + off); off = alignUp(off + (size_t)2*DD*DD*2, 256);   // q,k weight rows
    __bf16* wv  = (__bf16*)(ws + off); off = alignUp(off + (size_t)DD*DD*2, 256);
    __bf16* wo  = (__bf16*)(ws + off); off = alignUp(off + (size_t)DD*DD*2, 256);
    __bf16* qb  = (__bf16*)(ws + off); off = alignUp(off + (size_t)BB*HH*SS*DH*2, 256); // [B,H,S,DH]
    __bf16* kb  = (__bf16*)(ws + off); off = alignUp(off + (size_t)BB*HH*SS*DH*2, 256); // [B,H,S,DH]
    __bf16* vT  = (__bf16*)(ws + off); off = alignUp(off + (size_t)BB*HH*DH*SS*2, 256); // [B,H,DH,S]
    __bf16* oh  = (__bf16*)(ws + off); off = alignUp(off + (size_t)BB*SS*DD*2, 256);    // merged heads [B,S,D]
    float*  sc  = (float*)(ws + off);  off = alignUp(off + (size_t)BB*HH*SS*SS*4, 256); // scores / P (aliased)

    // --- Stage 0: fp32 -> bf16 conversions -------------------------------
    {
        int n;
        n = BB*SS*DD;  cvt_f32_bf16<<<(n+255)/256, 256, 0, stream>>>(x,    xb,  n);
        n = 2*DD*DD;   cvt_f32_bf16<<<(n+255)/256, 256, 0, stream>>>(in_w, wqk, n); // q,k rows only
        n = DD*DD;     cvt_f32_bf16<<<(n+255)/256, 256, 0, stream>>>(v_w,  wv,  n);
        n = DD*DD;     cvt_f32_bf16<<<(n+255)/256, 256, 0, stream>>>(out_w,wo,  n);
    }

    const int M4 = BB*SS;                     // 4096 rows of x
    // --- Stage 1: projections (Q, K head-split; Vc transposed) ----------
    {
        int tiles = (M4/16)*(DD/64);
        dim3 g((tiles+3)/4, 1);
        // Q: out index = b*(H*S*DH) + s*DH + h*(S*DH) + d
        gemm_bf16_wmma<<<g, 128, 0, stream>>>(xb, 0, DD, wqk, 0, DD, in_b, 1.0f,
            qb, 1, 1, 0, 0,
            SS, (size_t)HH*SS*DH, (size_t)DH,
            DH, (size_t)SS*DH, 1,
            M4, DD, DD);
        gemm_bf16_wmma<<<g, 128, 0, stream>>>(xb, 0, DD, wqk + (size_t)DD*DD, 0, DD, in_b + DD, 1.0f,
            kb, 1, 1, 0, 0,
            SS, (size_t)HH*SS*DH, (size_t)DH,
            DH, (size_t)SS*DH, 1,
            M4, DD, DD);
        // VcT: out index = b*(H*DH*S) + s + n*S   (n = h*DH+d)
        gemm_bf16_wmma<<<g, 128, 0, stream>>>(xb, 0, DD, wv, 0, DD, v_b, 1.0f,
            vT, 1, 1, 0, 0,
            SS, (size_t)HH*DH*SS, 1,
            1, (size_t)SS, 0,
            M4, DD, DD);
    }

    // --- Stage 2: scores = Q K^T / sqrt(DH), batched over B*H ----------
    {
        int tiles = (SS/16)*(SS/64);
        dim3 g((tiles+3)/4, BB*HH);
        gemm_bf16_wmma<<<g, 128, 0, stream>>>(
            qb, (size_t)SS*DH, DH, kb, (size_t)SS*DH, DH, nullptr, 0.125f,
            sc, 0, 1, (size_t)SS*SS, 0,
            1, (size_t)SS, 0,
            1, 1, 0,
            SS, SS, DH);
    }

    // --- Stage 3: fused softmax -> *M -> softmax -> bf16 P (in place) --
    {
        dim3 g(SS, BB*HH);
        softmax_mask_softmax<<<g, 256, 0, stream>>>(sc, Mmask);
    }

    // --- Stage 4: out_head = P @ Vc, merged-head store ------------------
    {
        __bf16* P = (__bf16*)sc;              // lda = 2*S (bf16 over f32 rows)
        int tiles = (SS/16)*(DH/64);
        dim3 g((tiles+3)/4, BB*HH);
        gemm_bf16_wmma<<<g, 128, 0, stream>>>(
            P, (size_t)SS*SS*2, 2*SS, vT, (size_t)DH*SS, SS, nullptr, 1.0f,
            oh, 1, HH, (size_t)SS*DD, (size_t)DH,
            1, (size_t)DD, 0,
            1, 1, 0,
            SS, DH, SS);
    }

    // --- Stage 5: final projection -> d_out (fp32) ----------------------
    {
        int tiles = (M4/16)*(DD/64);
        dim3 g((tiles+3)/4, 1);
        gemm_bf16_wmma<<<g, 128, 0, stream>>>(oh, 0, DD, wo, 0, DD, out_b, 1.0f,
            out, 0, 1, 0, 0,
            1, (size_t)DD, 0,
            1, 1, 0,
            M4, DD, DD);
    }
}
